// MultiHeadAttention_24962349924642
// MI455X (gfx1250) — matrix operand order repaired and checked by reading, not compiled, not run on hardware
//
#include <hip/hip_runtime.h>
#include <hip/hip_bf16.h>
#include <math.h>
#include <stdint.h>

// ---------------------------------------------------------------------------
// MI455X (gfx1250, wave32) fused MHA block:
//   qh/kh/vh = proj(q/k/v); S = qh khT / 8; P = softmax(S) (also an output);
//   ctx = P vh; x = ctx Wfc + b + residual; out = LayerNorm(x).
// GEMMs: v_wmma_f32_16x16x32_f16, 128x64 block tiles, 8 waves,
// 4 accumulators/wave (A fragment reused across 4 WMMAs).
// score_gemm stages its f16 tiles with GLOBAL_LOAD_ASYNC_TO_LDS_B64.
// ---------------------------------------------------------------------------

typedef __attribute__((ext_vector_type(16))) _Float16 v16h;
typedef __attribute__((ext_vector_type(8)))  float    v8f;
typedef __attribute__((ext_vector_type(4)))  _Float16 h4;

#define HN     8
#define DK     64
#define DM     512
#define BATCH  4
#define SEQ    2048
#define LN_EPS 1e-5f

// LDS tile strides in f16 elements: 40 halves = 80 B row (8-byte aligned rows)
#define LDA_S 40
#define LDB_S 40

// ---- WMMA fragment builders (ISA 7.12.2 layouts, wave32) ------------------
__device__ __forceinline__ v16h frag_a(const _Float16* ldsA, int row0, int lane) {
  const int m = row0 + (lane & 15);
  const int khsel = (lane >> 4) << 3;
  v16h a;
#pragma unroll
  for (int v = 0; v < 8; ++v) {
    const int kb = ((v >> 2) << 4) + khsel + ((v & 3) << 1);
    const _Float16* p = ldsA + m * LDA_S + kb;
    a[2 * v]     = p[0];
    a[2 * v + 1] = p[1];
  }
  return a;
}

__device__ __forceinline__ v16h frag_b(const _Float16* ldsB, int col0, int lane) {
  const int n  = col0 + (lane & 15);
  const int kg = (lane >> 4) << 4;
  v16h b;
#pragma unroll
  for (int v = 0; v < 8; ++v) {
    const _Float16* p = ldsB + n * LDB_S + ((v >> 2) << 4) + (kg >> 1) + ((v & 3) << 1);
    b[2 * v]     = p[0];
    b[2 * v + 1] = p[1];
  }
  return b;
}

__device__ __forceinline__ uint32_t lds_off(const void* p) {
  return (uint32_t)(uintptr_t)p;   // low 32 bits of generic ptr = LDS offset
}

// async f16 copy: 8 bytes/lane, global -> LDS, tracked by ASYNCcnt
__device__ __forceinline__ void async_copy_b64(const _Float16* gsrc, _Float16* ldst) {
  uint32_t dst = lds_off(ldst);
  asm volatile("global_load_async_to_lds_b64 %0, %1, off"
               :: "v"(dst), "v"(gsrc) : "memory");
}
__device__ __forceinline__ void async_wait() {
  asm volatile("s_wait_asynccnt 0x0" ::: "memory");
}

// ---------------------------------------------------------------------------
// Projection GEMM: C[M,N] = A[M,K](f32) * W[K,N](f32) + bias[N]
//   out16: write f16 head projections; out32: += residual, write f32 (pre-LN).
// Block tile 128x64, K chunk 32.
// ---------------------------------------------------------------------------
__global__ void __launch_bounds__(256)
proj_gemm(const float* __restrict__ A, int lda,
          const float* __restrict__ W, int ldw,
          const float* __restrict__ bias,
          const float* __restrict__ residual,
          _Float16* __restrict__ out16,
          float* __restrict__ out32,
          int M, int N, int K)
{
  __shared__ _Float16 ldsA[128 * LDA_S];
  __shared__ _Float16 ldsB[64 * LDB_S];
  const int tid = threadIdx.x, lane = tid & 31, wave = tid >> 5;
  const int n0 = blockIdx.x * 64, m0 = blockIdx.y * 128;
  const int row0 = wave * 16;

  v8f acc[4] = {{}, {}, {}, {}};

  for (int k0 = 0; k0 < K; k0 += 32) {
    // ---- stage A: 128x32 f32 -> f16, batched b128 loads, b64 LDS stores
    float4 fa[4];
#pragma unroll
    for (int i = 0; i < 4; ++i) {
      const int e = i * 256 + tid, r = e >> 3, c4 = (e & 7) << 2;
      fa[i] = *(const float4*)&A[(size_t)(m0 + r) * lda + (k0 + c4)];
    }
    // ---- stage W: 32x64 f32 -> f16 transposed (LDSB[n][k])
    float4 fb[2];
#pragma unroll
    for (int i = 0; i < 2; ++i) {
      const int e = i * 256 + tid, kk = e >> 4, n4 = (e & 15) << 2;
      fb[i] = *(const float4*)&W[(size_t)(k0 + kk) * ldw + (n0 + n4)];
    }
#pragma unroll
    for (int i = 0; i < 4; ++i) {
      const int e = i * 256 + tid, r = e >> 3, c4 = (e & 7) << 2;
      h4 h; h[0] = (_Float16)fa[i].x; h[1] = (_Float16)fa[i].y;
            h[2] = (_Float16)fa[i].z; h[3] = (_Float16)fa[i].w;
      *(h4*)&ldsA[r * LDA_S + c4] = h;
    }
#pragma unroll
    for (int i = 0; i < 2; ++i) {
      const int e = i * 256 + tid, kk = e >> 4, n4 = (e & 15) << 2;
      ldsB[(n4 + 0) * LDB_S + kk] = (_Float16)fb[i].x;
      ldsB[(n4 + 1) * LDB_S + kk] = (_Float16)fb[i].y;
      ldsB[(n4 + 2) * LDB_S + kk] = (_Float16)fb[i].z;
      ldsB[(n4 + 3) * LDB_S + kk] = (_Float16)fb[i].w;
    }
    __syncthreads();
    if (k0 + 32 < K)                              // -> global_prefetch_b8
      __builtin_prefetch(&A[(size_t)(m0 + (tid >> 1)) * lda + (k0 + 32)], 0, 1);

    const v16h a = frag_a(ldsA, row0, lane);      // reused across 4 WMMAs
#pragma unroll
    for (int c = 0; c < 4; ++c) {
      const v16h b = frag_b(ldsB, c * 16, lane);
      acc[c] = __builtin_amdgcn_wmma_f32_16x16x32_f16(false, a, false, b,
                                                      (short)0, acc[c], false, false);
    }
    __syncthreads();
  }

  const int nl = lane & 15, mb = (lane >> 4) << 3;
#pragma unroll
  for (int c = 0; c < 4; ++c) {
#pragma unroll
    for (int v = 0; v < 8; ++v) {
      const int mm = m0 + row0 + v + mb, nn = n0 + c * 16 + nl;
      float val = acc[c][v] + bias[nn];
      if (out32) { val += residual[(size_t)mm * N + nn];
                   out32[(size_t)mm * N + nn] = val; }
      else         out16[(size_t)mm * N + nn] = (_Float16)val;
    }
  }
}

// ---------------------------------------------------------------------------
// Scores: attn[h][b][lq][lk] = (qh[b,lq,h,:] . kh[b,lk,h,:]) / 8
// f16 tiles staged via GLOBAL_LOAD_ASYNC_TO_LDS_B64 (ASYNCcnt path).
// ---------------------------------------------------------------------------
__global__ void __launch_bounds__(256)
score_gemm(const _Float16* __restrict__ qh, const _Float16* __restrict__ kh,
           float* __restrict__ attn, float scale)
{
  __shared__ _Float16 ldsA[128 * LDA_S];
  __shared__ _Float16 ldsB[64 * LDB_S];
  const int tid = threadIdx.x, lane = tid & 31, wave = tid >> 5;
  const int n0 = blockIdx.x * 64, m0 = blockIdx.y * 128;
  const int h = blockIdx.z / BATCH, b = blockIdx.z % BATCH;
  const _Float16* qb = qh + (size_t)b * SEQ * DM + h * DK;
  const _Float16* kb = kh + (size_t)b * SEQ * DM + h * DK;
  const int row0 = wave * 16;

  v8f acc[4] = {{}, {}, {}, {}};

  for (int k0 = 0; k0 < DK; k0 += 32) {
#pragma unroll
    for (int i = 0; i < 4; ++i) {                 // A: 128x32 halves
      const int e = i * 256 + tid, r = e >> 3, c4 = (e & 7) << 2;
      async_copy_b64(&qb[(size_t)(m0 + r) * DM + (k0 + c4)],
                     &ldsA[r * LDA_S + c4]);
    }
#pragma unroll
    for (int i = 0; i < 2; ++i) {                 // B: LDSB[n][k] (rows of kh)
      const int e = i * 256 + tid, n = e >> 3, k4 = (e & 7) << 2;
      async_copy_b64(&kb[(size_t)(n0 + n) * DM + (k0 + k4)],
                     &ldsB[n * LDB_S + k4]);
    }
    async_wait();
    __syncthreads();

    const v16h a = frag_a(ldsA, row0, lane);
#pragma unroll
    for (int c = 0; c < 4; ++c) {
      const v16h b2 = frag_b(ldsB, c * 16, lane);
      acc[c] = __builtin_amdgcn_wmma_f32_16x16x32_f16(false, a, false, b2,
                                                      (short)0, acc[c], false, false);
    }
    __syncthreads();
  }

  float* ab = attn + (((size_t)h * BATCH + b) * SEQ) * SEQ;
  const int nl = lane & 15, mb = (lane >> 4) << 3;
#pragma unroll
  for (int c = 0; c < 4; ++c)
#pragma unroll
    for (int v = 0; v < 8; ++v)
      ab[(size_t)(m0 + row0 + v + mb) * SEQ + (n0 + c * 16 + nl)] = acc[c][v] * scale;
}

// ---------------------------------------------------------------------------
// Row softmax, in place, one 256-thread block per 2048-wide row (float4 I/O).
// ---------------------------------------------------------------------------
__global__ void __launch_bounds__(256)
softmax_rows(float* __restrict__ attn)
{
  __shared__ float red[256];
  float* p = attn + (size_t)blockIdx.x * SEQ;
  const int tid = threadIdx.x;

  float4 x0 = *(const float4*)&p[tid * 8];
  float4 x1 = *(const float4*)&p[tid * 8 + 4];
  float m = fmaxf(fmaxf(fmaxf(x0.x, x0.y), fmaxf(x0.z, x0.w)),
                  fmaxf(fmaxf(x1.x, x1.y), fmaxf(x1.z, x1.w)));
  red[tid] = m; __syncthreads();
  for (int s = 128; s > 0; s >>= 1) {
    if (tid < s) red[tid] = fmaxf(red[tid], red[tid + s]);
    __syncthreads();
  }
  const float rowmax = red[0];
  __syncthreads();

  x0.x = __expf(x0.x - rowmax); x0.y = __expf(x0.y - rowmax);
  x0.z = __expf(x0.z - rowmax); x0.w = __expf(x0.w - rowmax);
  x1.x = __expf(x1.x - rowmax); x1.y = __expf(x1.y - rowmax);
  x1.z = __expf(x1.z - rowmax); x1.w = __expf(x1.w - rowmax);
  red[tid] = (x0.x + x0.y + x0.z + x0.w) + (x1.x + x1.y + x1.z + x1.w);
  __syncthreads();
  for (int s = 128; s > 0; s >>= 1) {
    if (tid < s) red[tid] += red[tid + s];
    __syncthreads();
  }
  const float inv = 1.0f / red[0];
  x0.x *= inv; x0.y *= inv; x0.z *= inv; x0.w *= inv;
  x1.x *= inv; x1.y *= inv; x1.z *= inv; x1.w *= inv;
  *(float4*)&p[tid * 8]     = x0;
  *(float4*)&p[tid * 8 + 4] = x1;
}

// ---------------------------------------------------------------------------
// ctx[b, lq, h*64+dv] = sum_t attn[h,b,lq,t] * vh[b,t,h,dv]
// ---------------------------------------------------------------------------
__global__ void __launch_bounds__(256)
attn_v(const float* __restrict__ attn, const _Float16* __restrict__ vh,
       float* __restrict__ ctx)
{
  __shared__ _Float16 ldsA[128 * LDA_S];
  __shared__ _Float16 ldsB[64 * LDB_S];
  const int tid = threadIdx.x, lane = tid & 31, wave = tid >> 5;
  const int n0 = blockIdx.x * 64, m0 = blockIdx.y * 128;
  const int h = blockIdx.z / BATCH, b = blockIdx.z % BATCH;
  const float* ab = attn + (((size_t)h * BATCH + b) * SEQ) * SEQ;
  const _Float16* vb = vh + (size_t)b * SEQ * DM + h * DK;
  const int row0 = wave * 16;

  v8f acc[4] = {{}, {}, {}, {}};

  for (int k0 = 0; k0 < SEQ; k0 += 32) {
    float4 fa[4];
#pragma unroll
    for (int i = 0; i < 4; ++i) {                 // attn tile f32, batched b128
      const int e = i * 256 + tid, r = e >> 3, c4 = (e & 7) << 2;
      fa[i] = *(const float4*)&ab[(size_t)(m0 + r) * SEQ + (k0 + c4)];
    }
    h4 hb[2];
#pragma unroll
    for (int i = 0; i < 2; ++i) {                 // vh tile f16, rows of vh
      const int e = i * 256 + tid, kk = e >> 4, n4 = (e & 15) << 2;
      hb[i] = *(const h4*)&vb[(size_t)(k0 + kk) * DM + (n0 + n4)];
    }
#pragma unroll
    for (int i = 0; i < 4; ++i) {
      const int e = i * 256 + tid, r = e >> 3, c4 = (e & 7) << 2;
      h4 h; h[0] = (_Float16)fa[i].x; h[1] = (_Float16)fa[i].y;
            h[2] = (_Float16)fa[i].z; h[3] = (_Float16)fa[i].w;
      *(h4*)&ldsA[r * LDA_S + c4] = h;
    }
#pragma unroll
    for (int i = 0; i < 2; ++i) {                 // scatter-transpose into LDSB[n][k]
      const int e = i * 256 + tid, kk = e >> 4, n4 = (e & 15) << 2;
      ldsB[(n4 + 0) * LDB_S + kk] = hb[i][0];
      ldsB[(n4 + 1) * LDB_S + kk] = hb[i][1];
      ldsB[(n4 + 2) * LDB_S + kk] = hb[i][2];
      ldsB[(n4 + 3) * LDB_S + kk] = hb[i][3];
    }
    __syncthreads();
    if (k0 + 32 < SEQ)
      __builtin_prefetch(&ab[(size_t)(m0 + (tid >> 1)) * SEQ + (k0 + 32)], 0, 1);

    const v16h a = frag_a(ldsA, row0, lane);
#pragma unroll
    for (int c = 0; c < 4; ++c) {
      const v16h b2 = frag_b(ldsB, c * 16, lane);
      acc[c] = __builtin_amdgcn_wmma_f32_16x16x32_f16(false, a, false, b2,
                                                      (short)0, acc[c], false, false);
    }
    __syncthreads();
  }

  float* cb = ctx + (size_t)b * SEQ * DM + h * DK;
  const int nl = lane & 15, mb = (lane >> 4) << 3;
#pragma unroll
  for (int c = 0; c < 4; ++c)
#pragma unroll
    for (int v = 0; v < 8; ++v)
      cb[(size_t)(m0 + row0 + v + mb) * DM + (n0 + c * 16 + nl)] = acc[c][v];
}

// ---------------------------------------------------------------------------
// LayerNorm over last dim (512), one 256-thread block per row.
// ---------------------------------------------------------------------------
__global__ void __launch_bounds__(256)
layernorm(const float* __restrict__ x, const float* __restrict__ g,
          const float* __restrict__ bb, float* __restrict__ out)
{
  __shared__ float red[256];
  const float* px = x + (size_t)blockIdx.x * DM;
  float* po = out + (size_t)blockIdx.x * DM;
  const int tid = threadIdx.x;

  const float a = px[tid], c = px[tid + 256];
  red[tid] = a + c; __syncthreads();
  for (int s = 128; s > 0; s >>= 1) {
    if (tid < s) red[tid] += red[tid + s];
    __syncthreads();
  }
  const float mu = red[0] * (1.0f / DM);
  __syncthreads();

  const float da = a - mu, dc = c - mu;
  red[tid] = da * da + dc * dc; __syncthreads();
  for (int s = 128; s > 0; s >>= 1) {
    if (tid < s) red[tid] += red[tid + s];
    __syncthreads();
  }
  const float inv = rsqrtf(red[0] * (1.0f / DM) + LN_EPS);
  po[tid]       = da * inv * g[tid] + bb[tid];
  po[tid + 256] = dc * inv * g[tid + 256] + bb[tid + 256];
}

// ---------------------------------------------------------------------------
extern "C" void kernel_launch(void* const* d_in, const int* in_sizes, int n_in,
                              void* d_out, int out_size, void* d_ws, size_t ws_size,
                              hipStream_t stream) {
  const float* q    = (const float*)d_in[0];
  const float* k    = (const float*)d_in[1];
  const float* v    = (const float*)d_in[2];
  const float* w_q  = (const float*)d_in[3];
  const float* b_q  = (const float*)d_in[4];
  const float* w_k  = (const float*)d_in[5];
  const float* b_k  = (const float*)d_in[6];
  const float* w_v  = (const float*)d_in[7];
  const float* b_v  = (const float*)d_in[8];
  const float* w_fc = (const float*)d_in[9];
  const float* b_fc = (const float*)d_in[10];
  const float* ln_g = (const float*)d_in[11];
  const float* ln_b = (const float*)d_in[12];

  const size_t tok = (size_t)BATCH * SEQ;      // 8192 rows
  float* out_ln = (float*)d_out;
  float* attn   = out_ln + tok * DM;           // [H,B,L,L] region of d_out

  // workspace: qh|kh|vh (f16) then ctx (f32); x aliases qh+kh (dead by then)
  _Float16* qh = (_Float16*)d_ws;
  _Float16* kh = qh + tok * DM;
  _Float16* vh = kh + tok * DM;
  float*    ctx = (float*)(vh + tok * DM);
  float*    x   = (float*)d_ws;

  const dim3 blk(256);
  const dim3 gproj(DM / 64, (unsigned)(tok / 128));

  proj_gemm<<<gproj, blk, 0, stream>>>(q, DM, w_q, DM, b_q, nullptr,
                                       qh, nullptr, (int)tok, DM, DM);
  proj_gemm<<<gproj, blk, 0, stream>>>(k, DM, w_k, DM, b_k, nullptr,
                                       kh, nullptr, (int)tok, DM, DM);
  proj_gemm<<<gproj, blk, 0, stream>>>(v, DM, w_v, DM, b_v, nullptr,
                                       vh, nullptr, (int)tok, DM, DM);

  const dim3 gscore(SEQ / 64, SEQ / 128, HN * BATCH);
  score_gemm<<<gscore, blk, 0, stream>>>(qh, kh, attn, 0.125f);

  softmax_rows<<<dim3(HN * BATCH * SEQ), blk, 0, stream>>>(attn);

  const dim3 gav(1, SEQ / 128, HN * BATCH);
  attn_v<<<gav, blk, 0, stream>>>(attn, vh, ctx);

  proj_gemm<<<gproj, blk, 0, stream>>>(ctx, DM, w_fc, DM, b_fc, q,
                                       nullptr, x, (int)tok, DM, DM);

  layernorm<<<dim3((unsigned)tok), blk, 0, stream>>>(x, ln_g, ln_b, out_ln);
}
